// BertCrossAttention_80685255623404
// MI455X (gfx1250) — matrix operand order repaired and checked by reading, not compiled, not run on hardware
//
#include <hip/hip_runtime.h>

typedef __bf16 bf16;
typedef __attribute__((ext_vector_type(16))) __bf16 v16bf;
typedef __attribute__((ext_vector_type(8)))  __bf16 v8bf;
typedef __attribute__((ext_vector_type(8)))  float  v8f;

#define WAVES_PER_BLOCK 8
#define BLOCK_THREADS 256

__device__ __forceinline__ v8f wmma_bf16(v16bf a, v16bf b, v8f c) {
  return __builtin_amdgcn_wmma_f32_16x16x32_bf16(false, a, false, b, (short)0, c,
                                                 false, false);
}

__device__ __forceinline__ v16bf bf_join(v8bf lo, v8bf hi) {
  v16bf r;
#pragma unroll
  for (int i = 0; i < 8; i++) { r[i] = lo[i]; r[8 + i] = hi[i]; }
  return r;
}

// A-fragment (16x32 bf16). rowptr = this lane's row (M = lane%16), kc = 0 or 32.
// Lanes 0-15 hold K = kc+{0..15}, lanes 16-31 K = kc+{16..31}.
__device__ __forceinline__ v16bf load_afrag(const bf16* rowptr, int kc, int half) {
  const bf16* p = rowptr + kc + (half ? 16 : 0);
  return bf_join(*(const v8bf*)p, *(const v8bf*)(p + 8));
}

__device__ __forceinline__ v16bf load_afrag_f32(const float* rowptr, int kc, int half) {
  const float* p = rowptr + kc + (half ? 16 : 0);
  v16bf r;
#pragma unroll
  for (int i = 0; i < 8; i++) { r[i] = (__bf16)p[i]; r[8 + i] = (__bf16)p[8 + i]; }
  return r;
}

// B-fragment (32x16 bf16) where GEMM-B is the transpose of a row-major [N,K] source.
// p = &src[n_row][kbase], kbase = kc + (half?16:0); 16 contiguous bf16 per lane.
__device__ __forceinline__ v16bf load_bfragT(const bf16* p) {
  return bf_join(*(const v8bf*)p, *(const v8bf*)(p + 8));
}

__device__ __forceinline__ v16bf load_bfragT_f32(const float* p) {
  v16bf r;
#pragma unroll
  for (int i = 0; i < 16; i++) r[i] = (__bf16)p[i];
  return r;
}

// out = X @ W^T + bias, stored as bf16.
// mode 0: Q  -> [b,h, l, d] (Lctx=1024); mode 1: K -> [b,h, r, d] (2048);
// mode 2: V^T-> [b,h, d, r] (2048)
__global__ __launch_bounds__(BLOCK_THREADS)
void proj_kernel(const float* __restrict__ X, const float* __restrict__ W,
                 const float* __restrict__ bias, bf16* __restrict__ out,
                 int Lctx, int mode) {
  const int tid  = threadIdx.x;
  const int lane = tid & 31, wid = tid >> 5;
  const int half = lane >> 4, lrow = lane & 15;
  const int w     = blockIdx.x * WAVES_PER_BLOCK + wid;
  const int tileN = w & 15;
  const int tileM = w >> 4;
  const int m0 = tileM * 16, n0 = tileN * 64;

  const float* xrow = X + (size_t)(m0 + lrow) * 1024;
  v8f acc[4] = {};

  for (int kk = 0; kk < 1024; kk += 32) {
    v16bf a = load_afrag_f32(xrow, kk, half);
#pragma unroll
    for (int c = 0; c < 4; c++) {
      const float* wp = W + (size_t)(n0 + c * 16 + lrow) * 1024 + kk + (half ? 16 : 0);
      acc[c] = wmma_bf16(a, load_bfragT_f32(wp), acc[c]);
    }
  }

#pragma unroll
  for (int c = 0; c < 4; c++) {
    const int n_g = n0 + c * 16 + lrow;
    const float bv = bias[n_g];
    const int h = n_g >> 6, d = n_g & 63;
#pragma unroll
    for (int v = 0; v < 8; v++) {
      const int row = m0 + v + (half ? 8 : 0);
      const int b = row / Lctx, l = row % Lctx;
      const float val = acc[c][v] + bv;
      size_t idx;
      if (mode == 2) idx = ((size_t)((b * 16 + h) * 64 + d)) * Lctx + l;
      else           idx = ((size_t)(b * 16 + h) * Lctx + l) * 64 + d;
      out[idx] = (__bf16)val;
    }
  }
}

// Flash attention with relative_key_query band terms.
// 1 wave = one (b,h, 16-row q block); 64 iterations of 32 k-columns.
// Band chunks (16 emb rows each): t0 = l0-r0+2016; chunk c covers t0+16c.
// Next iteration's chunk2 == this iteration's chunk0 (carried in registers).
__global__ __launch_bounds__(BLOCK_THREADS)
void attn_kernel(const bf16* __restrict__ Q, const bf16* __restrict__ Km,
                 const bf16* __restrict__ VT, const float* __restrict__ emb,
                 const float* __restrict__ mask, float* __restrict__ out) {
  __shared__ float Gl[WAVES_PER_BLOCK][16][49];      // 3 chunks of 16 j-cols
  __shared__ float Hl[WAVES_PER_BLOCK][2][16][33];   // per-subtile 31-wide band
  __shared__ float Pl[WAVES_PER_BLOCK][16][33];      // probs 16x32

  const int tid  = threadIdx.x;
  const int lane = tid & 31, wid = tid >> 5;
  const int half = lane >> 4, lrow = lane & 15;
  const int w  = blockIdx.x * WAVES_PER_BLOCK + wid;
  const int bh = w >> 6;
  const int l0 = (w & 63) << 4;
  const int b = bh >> 4, h = bh & 15;
  const int kb16 = half ? 16 : 0;

  const bf16* qrow = Q + ((size_t)bh * 1024 + l0 + lrow) * 64;
  const v16bf aq0 = load_afrag(qrow, 0, half);
  const v16bf aq1 = load_afrag(qrow, 32, half);

  // init carried chunk2 (emb rows l0+2048 .. l0+2063, max 3071 < 4095)
  const float* e2 = emb + (size_t)(l0 + 2048 + lrow) * 64 + kb16;
  v16bf ebc0 = load_bfragT_f32(e2);
  v16bf ebc1 = load_bfragT_f32(e2 + 32);
  v8f Gc = {};
  Gc = wmma_bf16(aq0, ebc0, Gc);
  Gc = wmma_bf16(aq1, ebc1, Gc);

  float m_i[8], l_i[8];
  v8f acc[4] = {};
#pragma unroll
  for (int v = 0; v < 8; v++) { m_i[v] = -__builtin_inff(); l_i[v] = 0.f; }

  const float* maskrow = mask + (size_t)b * 2048;

  for (int r0 = 0; r0 < 2048; r0 += 32) {
    const int t0 = l0 - r0 + 2016;                 // >= 0, chunk rows <= 3071

    // ---- new band chunks 0,1 and their G tiles (chunk2 carried) ----
    const float* e0 = emb + (size_t)(t0 + lrow) * 64 + kb16;
    const float* e1 = emb + (size_t)(t0 + 16 + lrow) * 64 + kb16;
    const v16bf eb0_0 = load_bfragT_f32(e0), eb0_1 = load_bfragT_f32(e0 + 32);
    const v16bf eb1_0 = load_bfragT_f32(e1), eb1_1 = load_bfragT_f32(e1 + 32);
    v8f G0 = {}, G1 = {};
    G0 = wmma_bf16(aq0, eb0_0, G0); G0 = wmma_bf16(aq1, eb0_1, G0);
    G1 = wmma_bf16(aq0, eb1_0, G1); G1 = wmma_bf16(aq1, eb1_1, G1);

#pragma unroll
    for (int v = 0; v < 8; v++) {
      const int m = v + (half ? 8 : 0);
      Gl[wid][m][lrow]      = G0[v];
      Gl[wid][m][16 + lrow] = G1[v];
      Gl[wid][m][32 + lrow] = Gc[v];
    }

    // ---- per 16-col subtile: QK^T and H band tiles ----
    v8f s[2];
#pragma unroll
    for (int ss = 0; ss < 2; ss++) {
      const bf16* krow = Km + ((size_t)bh * 2048 + r0 + 16 * ss + lrow) * 64;
      v8f t = {};
      t = wmma_bf16(aq0, load_bfragT(krow + kb16), t);
      t = wmma_bf16(aq1, load_bfragT(krow + 32 + kb16), t);
      s[ss] = t;

      const v16bf ak0 = load_afrag(krow, 0, half);
      const v16bf ak1 = load_afrag(krow, 32, half);
      // subtile 0: local band = chunks {1,2}; subtile 1: chunks {0,1}
      v8f hlo = {}, hhi = {};
      if (ss == 0) {
        hlo = wmma_bf16(ak0, eb1_0, hlo); hlo = wmma_bf16(ak1, eb1_1, hlo);
        hhi = wmma_bf16(ak0, ebc0, hhi);  hhi = wmma_bf16(ak1, ebc1, hhi);
      } else {
        hlo = wmma_bf16(ak0, eb0_0, hlo); hlo = wmma_bf16(ak1, eb0_1, hlo);
        hhi = wmma_bf16(ak0, eb1_0, hhi); hhi = wmma_bf16(ak1, eb1_1, hhi);
      }
#pragma unroll
      for (int v = 0; v < 8; v++) {
        const int m = v + (half ? 8 : 0);
        Hl[wid][ss][m][lrow]      = hlo[v];
        Hl[wid][ss][m][16 + lrow] = hhi[v];
      }
    }
    __asm__ volatile("" ::: "memory");

    // ---- diagonal gather + scale + mask ----
#pragma unroll
    for (int ss = 0; ss < 2; ss++)
#pragma unroll
      for (int v = 0; v < 8; v++) {
        const int m = v + (half ? 8 : 0);
        const int j = m + 15 - lrow;                       // [0,30] within subtile
        float sv = s[ss][v] + Gl[wid][m][j + 16 - 16 * ss] + Hl[wid][ss][lrow][j];
        s[ss][v] = sv * 0.125f + maskrow[r0 + 16 * ss + lrow];
      }

    // ---- online softmax over 32 columns ----
#pragma unroll
    for (int v = 0; v < 8; v++) {
      float rm = fmaxf(s[0][v], s[1][v]);
#pragma unroll
      for (int off = 8; off; off >>= 1) rm = fmaxf(rm, __shfl_xor(rm, off, 32));
      const float mnew = fmaxf(m_i[v], rm);
      const float p0 = __expf(s[0][v] - mnew);
      const float p1 = __expf(s[1][v] - mnew);
      float rs = p0 + p1;
#pragma unroll
      for (int off = 8; off; off >>= 1) rs += __shfl_xor(rs, off, 32);
      const float alpha = __expf(m_i[v] - mnew);
      l_i[v] = l_i[v] * alpha + rs;
      m_i[v] = mnew;
      s[0][v] = p0; s[1][v] = p1;
#pragma unroll
      for (int c = 0; c < 4; c++) acc[c][v] *= alpha;
    }

    // ---- P (16x32, full K) @ V via V^T ----
    __asm__ volatile("" ::: "memory");
#pragma unroll
    for (int ss = 0; ss < 2; ss++)
#pragma unroll
      for (int v = 0; v < 8; v++)
        Pl[wid][v + (half ? 8 : 0)][16 * ss + lrow] = s[ss][v];
    __asm__ volatile("" ::: "memory");

    v16bf ap;
    {
      const int kb = half ? 16 : 0;
#pragma unroll
      for (int i = 0; i < 8; i++) {
        ap[i]     = (__bf16)Pl[wid][lrow][kb + i];
        ap[8 + i] = (__bf16)Pl[wid][lrow][kb + 8 + i];
      }
    }
#pragma unroll
    for (int c = 0; c < 4; c++) {
      const bf16* vp = VT + ((size_t)bh * 64 + c * 16 + lrow) * 2048 + r0 + kb16;
      acc[c] = wmma_bf16(ap, load_bfragT(vp), acc[c]);
    }

    // ---- carry: current chunk0 becomes next iteration's chunk2 ----
    ebc0 = eb0_0; ebc1 = eb0_1; Gc = G0;
  }

  // ---- epilogue: ctx / l_i, write [B, Lq, H*64] fp32 ----
#pragma unroll
  for (int v = 0; v < 8; v++) {
    const int m = v + (half ? 8 : 0);
    const float inv = 1.f / l_i[v];
    const size_t row = (size_t)b * 1024 + l0 + m;
#pragma unroll
    for (int c = 0; c < 4; c++)
      out[row * 1024 + h * 64 + c * 16 + lrow] = acc[c][v] * inv;
  }
}

extern "C" void kernel_launch(void* const* d_in, const int* in_sizes, int n_in,
                              void* d_out, int out_size, void* d_ws, size_t ws_size,
                              hipStream_t stream) {
  const float* hidden  = (const float*)d_in[0];  // [4,2048,1024]
  const float* qhidden = (const float*)d_in[1];  // [4,1024,1024]
  const float* mask    = (const float*)d_in[2];  // [4,1,1,2048]
  const float* Wq = (const float*)d_in[3];
  const float* bq = (const float*)d_in[4];
  const float* Wk = (const float*)d_in[5];
  const float* bk = (const float*)d_in[6];
  const float* Wv = (const float*)d_in[7];
  const float* bv = (const float*)d_in[8];
  const float* emb = (const float*)d_in[9];      // [4095, 64]

  char* ws = (char*)d_ws;
  bf16* Qbf  = (bf16*)ws;                                            // 8 MB
  bf16* Kbf  = (bf16*)(ws + (size_t)64 * 1024 * 64 * 2);             // 16 MB
  bf16* VTbf = (bf16*)(ws + (size_t)64 * 1024 * 64 * 2
                          + (size_t)64 * 2048 * 64 * 2);             // 16 MB

  proj_kernel<<<4096 / 8, BLOCK_THREADS, 0, stream>>>(qhidden, Wq, bq, Qbf, 1024, 0);
  proj_kernel<<<8192 / 8, BLOCK_THREADS, 0, stream>>>(hidden, Wk, bk, Kbf, 2048, 1);
  proj_kernel<<<8192 / 8, BLOCK_THREADS, 0, stream>>>(hidden, Wv, bv, VTbf, 2048, 2);
  attn_kernel<<<4096 / 8, BLOCK_THREADS, 0, stream>>>(Qbf, Kbf, VTbf, emb, mask,
                                                      (float*)d_out);
}
